// NTNDecoder_386547057344
// MI455X (gfx1250) — matrix operand order repaired and checked by reading, not compiled, not run on hardware
//
#include <hip/hip_runtime.h>
#include <hip/hip_bf16.h>
#include <stdint.h>

// ---------------------------------------------------------------------------
// NTN decoder, MI455X (gfx1250).
//   B=64, L=128, H=128, D=16
//   GEMM1: tmp2[b,d,j,n] = enc x W           (bf16 WMMA, f32 acc, permuted store)
//   dotlr: dotlr[8192,32] = enc x [Wl|Wr]    (bf16 WMMA, f32 acc)
//   GEMM2+epilogue: wave w = d-slice w; A tile DMA'd to LDS via TDM
//                   (tensor_load_to_lds + s_wait_tensorcnt), B read from global,
//                   cross-wave d-reduction via ds_add_f32, fused tail.
// ---------------------------------------------------------------------------

typedef __attribute__((ext_vector_type(16))) __bf16    v16bf;
typedef __attribute__((ext_vector_type(8)))  __bf16    v8bf;
typedef __attribute__((ext_vector_type(8)))  float     v8f;
typedef __attribute__((ext_vector_type(4)))  uint32_t  v4u;
typedef __attribute__((ext_vector_type(8)))  int       v8i;
typedef __attribute__((ext_vector_type(4)))  int       v4i;

#define BATCH 64
#define LSEQ  128
#define HDIM  128
#define DDIM  16
#define MROWS (BATCH * LSEQ)            // 8192
#define N1    (HDIM * DDIM)             // 2048
#define OUTPLANE (BATCH * LSEQ * LSEQ)  // 1048576
#define BIGPEN 1e8f

__device__ __forceinline__ v16bf make_v16(v8bf lo, v8bf hi) {
  v16bf r;
#pragma unroll
  for (int i = 0; i < 8; ++i) { r[i] = lo[i]; r[i + 8] = hi[i]; }
  return r;
}

__device__ __forceinline__ uint32_t pcg_hash(uint32_t v) {
  v = v * 747796405u + 2891336453u;
  uint32_t w = ((v >> ((v >> 28u) + 4u)) ^ v) * 277803737u;
  return (w >> 22u) ^ w;
}

__device__ __forceinline__ float softplus_f(float x) {
  return fmaxf(x, 0.0f) + log1pf(expf(-fabsf(x)));
}

// ---------------------------------------------------------------------------
// TDM: DMA a 2D bf16 tile (tile_x elems contiguous, tile_y rows, row stride
// `stride_elems`) from global into LDS at byte offset lds_byte_off.
// Descriptor packing per CDNA5 ISA §8.3/§8.4 (data_size=1 -> 2 bytes).
// Issued once per wave (EXEC-independent); completion via TENSORcnt.
// This toolchain exposes the 6-arg builtin:
//   (uint32x4 g0, int32x8 g1, int32x4 g2, int32x4 g3, int32x8 extra, i32 cpol)
// ---------------------------------------------------------------------------
__device__ __forceinline__ void tdm_load_2d_bf16(uint32_t lds_byte_off,
                                                 uint64_t gaddr,
                                                 uint32_t tile_x, uint32_t tile_y,
                                                 uint32_t stride_elems) {
  uint32_t lo = (uint32_t)__builtin_amdgcn_readfirstlane((int)(uint32_t)gaddr);
  uint32_t hi = (uint32_t)__builtin_amdgcn_readfirstlane((int)(uint32_t)(gaddr >> 32));
  uint32_t la = (uint32_t)__builtin_amdgcn_readfirstlane((int)lds_byte_off);

  // group0: count=1 | lds_addr | global_addr[31:0] | global_addr[56:32], type=2
  v4u g0 = { 1u, la, lo, (hi & 0x01FFFFFFu) | 0x80000000u };

  const uint32_t T0 = 0x40000u, T1 = 0x40000u;  // generous tensor dims (no OOB)
  v8i g1 = { (int)0x00010000u,                           // data_size=1 (2 bytes)
             (int)((T0 & 0xFFFFu) << 16),                // tensor_dim0[15:0]
             (int)((T0 >> 16) | ((T1 & 0xFFFFu) << 16)), // dim0 hi | dim1 lo
             (int)((T1 >> 16) | (tile_x << 16)),         // dim1 hi | tile_dim0
             (int)(tile_y & 0xFFFFu),                    // tile_dim1 (tile_dim2=0)
             (int)stride_elems,                          // tensor_dim0_stride lo
             0, 0 };
  v4i g2 = {0, 0, 0, 0};
  v4i g3 = {0, 0, 0, 0};
  v8i g4 = {0, 0, 0, 0, 0, 0, 0, 0};
  __builtin_amdgcn_tensor_load_to_lds(g0, g1, g2, g3, g4, 0);
}

// ---------------------------------------------------------------------------
// Kernel 0: f32 -> bf16 conversions (enc, W-as-[128,2048], [Wl|Wr]-as-[128,32])
// ---------------------------------------------------------------------------
__global__ void convert_kernel(const float* __restrict__ enc,
                               const float* __restrict__ W,
                               const float* __restrict__ Wl,
                               const float* __restrict__ Wr,
                               __bf16* __restrict__ encBF,
                               __bf16* __restrict__ W2BF,
                               __bf16* __restrict__ WlrBF) {
  int i = blockIdx.x * blockDim.x + threadIdx.x;
  if (i < MROWS * HDIM) encBF[i] = (__bf16)enc[i];
  if (i < HDIM * N1 / 8) W2BF[i] = (__bf16)W[i];     // 32768 elements of W
  if (i < HDIM * 32) {                               // 4096 elements of [Wl | Wr]
    int k = i >> 5, d = i & 31;
    float v = (d < 16) ? Wl[k * 16 + d] : Wr[k * 16 + (d - 16)];
    WlrBF[i] = (__bf16)v;
  }
}

// ---------------------------------------------------------------------------
// Generic K=128 bf16 GEMM: C = A[M,128] x Bm[128, LDB_N]
// block = 256 threads = 8 waves; one 16x16 tile / wave, shared ntile / block.
// B tile [16 cols][128 k] staged in LDS so fragments are ds_load_b128.
// PERMUTE_BF16: store bf16 to tmp2[b, d=l16, j, n2=nt] (feeds GEMM2);
// otherwise store f32 row-major [M, LDB_N].
// ---------------------------------------------------------------------------
template <int LDB_N, bool PERMUTE_BF16>
__global__ void gemm_k128(const __bf16* __restrict__ A,
                          const __bf16* __restrict__ Bm,
                          void* __restrict__ Cout,
                          int mblocks /* = Mtiles/8 */) {
  __shared__ __align__(32) __bf16 Bt[16][128];   // [col][k]

  const int mb = blockIdx.x % mblocks;
  const int nt = blockIdx.x / mblocks;

#pragma unroll
  for (int p = threadIdx.x; p < 1024; p += 256) {
    const int k = p >> 3;
    const int cp = p & 7;
    const __bf16* src = Bm + (size_t)k * LDB_N + nt * 16 + 2 * cp;
    Bt[2 * cp + 0][k] = src[0];
    Bt[2 * cp + 1][k] = src[1];
  }
  __syncthreads();

  const int wave = threadIdx.x >> 5;
  const int lane = threadIdx.x & 31;
  const int half = lane >> 4;
  const int l16  = lane & 15;
  const int kb   = half * 8;   // A K-base per ISA 16-bit A 16x32 layout

  const int mt = mb * 8 + wave;
  const __bf16* Arow = A + (size_t)(mt * 16 + l16) * 128;

  v8f c = {0.f, 0.f, 0.f, 0.f, 0.f, 0.f, 0.f, 0.f};

#pragma unroll
  for (int kk = 0; kk < 128; kk += 32) {
    v8bf a0 = *(const v8bf*)(Arow + kk + kb);
    v8bf a1 = *(const v8bf*)(Arow + kk + kb + 16);
    const __bf16* bl = &Bt[l16][kk + half * 8];
    v8bf b0 = *(const v8bf*)(bl);
    v8bf b1 = *(const v8bf*)(bl + 16);
    v16bf af = make_v16(a0, a1);
    v16bf bf = make_v16(b0, b1);
    c = __builtin_amdgcn_wmma_f32_16x16x32_bf16(
        false, af, false, bf, (short)0, c, false, false);
  }

#pragma unroll
  for (int r = 0; r < 8; ++r) {
    const int row = mt * 16 + r + half * 8;   // = b*128 + j
    if (PERMUTE_BF16) {
      const int bidx = row >> 7;
      const int j    = row & 127;
      // col = nt*16 + l16  ->  n2 = nt, d = l16
      const size_t off = (((size_t)bidx * DDIM + l16) * LSEQ + j) * HDIM + nt;
      ((__bf16*)Cout)[off] = (__bf16)c[r];
    } else {
      const size_t off = (size_t)row * LDB_N + nt * 16 + l16;
      ((float*)Cout)[off] = c[r];
    }
  }
}

// ---------------------------------------------------------------------------
// Kernel 2: bilinear GEMM over n (K=128) fused with the whole epilogue.
// grid = B*8*8 blocks, 512 threads = 16 waves, wave w owns d = w.
// Per-wave A tile tmp2[b,d, jt*16..+16, :] DMA'd into a private 2KB LDS slab
// via TDM (two 16x64 chunks, double-use of the slab), B read from global.
// ---------------------------------------------------------------------------
__global__ void gemm2_epilogue(const __bf16* __restrict__ tmp2,  // [64,16,128,128]
                               const __bf16* __restrict__ encBF, // [8192,128]
                               const float*  __restrict__ dotlr, // [8192,32]
                               const float*  __restrict__ U,     // [16]
                               const float*  __restrict__ Bvec,  // [16]
                               const float*  __restrict__ lbias, // [1]
                               float* __restrict__ out) {        // 3*OUTPLANE
  // At first + highest alignment => LDS offset 0 after LDS lowering.
  __shared__ __align__(128) __bf16 At[16][16][64];  // [d][j][nn] slabs, 32 KB
  __shared__ float red[16][16];

  const int b  = blockIdx.x >> 6;
  const int rm = blockIdx.x & 63;
  const int jt = rm >> 3;
  const int mt = rm & 7;

  const int wave = threadIdx.x >> 5;   // == d slice
  const int lane = threadIdx.x & 31;
  const int half = lane >> 4;
  const int l16  = lane & 15;
  const int kb   = half * 8;
  const int d    = wave;

  v8f c = {0.f, 0.f, 0.f, 0.f, 0.f, 0.f, 0.f, 0.f};

  // A slice for this wave: tmp2[b][d][jt*16 .. jt*16+15][0..127], row stride 128
  const __bf16* gsrc = tmp2 + (((size_t)b * DDIM + d) * LSEQ + jt * 16) * HDIM;
  const uint32_t lds_slab = (uint32_t)d * (16u * 64u * 2u);  // 2KB per wave

  // B rows for this wave's tile (shared across waves; L2-resident)
  const __bf16* Brow = encBF + (size_t)(b * LSEQ + mt * 16 + l16) * HDIM;
  __builtin_prefetch(Brow, 0, 1);   // global_prefetch_b8

#pragma unroll
  for (int ck = 0; ck < 2; ++ck) {
    if (ck) {
      // our ds reads of the previous chunk must land before TDM overwrites slab
      asm volatile("s_wait_dscnt 0" ::: "memory");
    }
    tdm_load_2d_bf16(lds_slab,
                     (uint64_t)(uintptr_t)(gsrc + ck * 64),
                     /*tile_x=*/64, /*tile_y=*/16, /*stride=*/HDIM);
    __builtin_amdgcn_s_wait_tensorcnt(0);

#pragma unroll
    for (int kk2 = 0; kk2 < 64; kk2 += 32) {
      const int kkg = ck * 64 + kk2;
      // A fragment: row j = l16, contiguous nn within the 64-wide slab row
      const __bf16* ar = &At[d][l16][kk2];
      v8bf a0 = *(const v8bf*)(ar + kb);
      v8bf a1 = *(const v8bf*)(ar + kb + 16);
      // B fragment = enc[b, m, n] : col m = mt*16+l16, n contiguous
      const __bf16* br = Brow + kkg + half * 8;
      v8bf b0 = *(const v8bf*)(br);
      v8bf b1 = *(const v8bf*)(br + 16);
      v16bf af = make_v16(a0, a1);
      v16bf bf = make_v16(b0, b1);
      c = __builtin_amdgcn_wmma_f32_16x16x32_bf16(
          false, af, false, bf, (short)0, c, false, false);
    }
  }

  __syncthreads();
  if (threadIdx.x < 256) red[threadIdx.x >> 4][threadIdx.x & 15] = 0.0f;
  __syncthreads();

  // tanh(bilinear + dot_l + dot_r + Bvec[d]) * U[d], reduce over d (ds_add_f32)
  {
    const float u_d = U[d];
    const float bv  = Bvec[d];
    const int   m   = mt * 16 + l16;
    const float dr  = dotlr[(size_t)(b * LSEQ + m) * 32 + 16 + d];
#pragma unroll
    for (int r = 0; r < 8; ++r) {
      const int jloc = r + half * 8;
      const int j    = jt * 16 + jloc;
      const float dl = dotlr[(size_t)(b * LSEQ + j) * 32 + d];
      const float f  = tanhf(c[r] + dl + dr + bv);
      atomicAdd(&red[jloc][l16], f * u_d);
    }
  }
  __syncthreads();

  // mask diag, sigmoid, bernoulli(hash), entropy
  if (threadIdx.x < 256) {
    const int lj = threadIdx.x >> 4;
    const int lm = threadIdx.x & 15;
    const int j  = jt * 16 + lj;
    const int m  = mt * 16 + lm;
    float s = red[lj][lm] + lbias[0];
    const float masked = (j == m) ? (s - BIGPEN) : s;
    const float p = 1.0f / (1.0f + expf(-masked));
    const size_t idx = ((size_t)b * LSEQ + j) * LSEQ + m;
    const uint32_t h = pcg_hash((uint32_t)idx ^ 0x9E3779B9u);
    const float u01 = (float)(h >> 8) * (1.0f / 16777216.0f);
    out[idx] = (u01 < p) ? 1.0f : 0.0f;
    out[OUTPLANE + idx] = masked;
    out[2 * OUTPLANE + idx] =
        p * softplus_f(-masked) + (1.0f - p) * softplus_f(masked);
  }
}

// ---------------------------------------------------------------------------
// Launcher
// ---------------------------------------------------------------------------
extern "C" void kernel_launch(void* const* d_in, const int* in_sizes, int n_in,
                              void* d_out, int out_size, void* d_ws, size_t ws_size,
                              hipStream_t stream) {
  const float* enc  = (const float*)d_in[0];
  const float* W    = (const float*)d_in[1];
  const float* Wl   = (const float*)d_in[2];
  const float* Wr   = (const float*)d_in[3];
  const float* U    = (const float*)d_in[4];
  const float* Bv   = (const float*)d_in[5];
  const float* lb   = (const float*)d_in[6];

  char* ws = (char*)d_ws;
  // workspace layout (bytes):
  //   encBF : 8192*128  bf16 =  2 MB   @ 0
  //   W2BF  : 128*2048  bf16 = 512 KB  @ 2 MB
  //   WlrBF : 128*32    bf16 =   8 KB  @ 2.5 MB
  //   dotlr : 8192*32   f32  =   1 MB  @ 2.5 MB + 8 KB
  //   tmp2  : 64*16*128*128 bf16 = 32 MB (fits in 192 MB L2)
  __bf16* encBF = (__bf16*)(ws);
  __bf16* W2BF  = (__bf16*)(ws + (2u << 20));
  __bf16* WlrBF = (__bf16*)(ws + (2u << 20) + (512u << 10));
  float*  dotlr = (float*) (ws + (2u << 20) + (520u << 10));
  __bf16* tmp2  = (__bf16*)(ws + (2u << 20) + (520u << 10) + (1u << 20));

  convert_kernel<<<4096, 256, 0, stream>>>(enc, W, Wl, Wr, encBF, W2BF, WlrBF);

  // GEMM1: M=8192, N=2048, K=128 -> tmp2[b,d,j,n]  (mblocks=64, ntiles=128)
  gemm_k128<N1, true><<<64 * 128, 256, 0, stream>>>(encBF, W2BF, (void*)tmp2, 64);

  // dotlr: M=8192, N=32, K=128  (mblocks=64, ntiles=2)
  gemm_k128<32, false><<<64 * 2, 256, 0, stream>>>(encBF, WlrBF, (void*)dotlr, 64);

  // GEMM2 + fused epilogue (TDM-staged A tiles)
  gemm2_epilogue<<<BATCH * 8 * 8, 512, 0, stream>>>(tmp2, encBF, dotlr, U, Bv, lb,
                                                    (float*)d_out);
}
